// ATMASKDGCNN_10728828305705
// MI455X (gfx1250) — matrix operand order repaired and checked by reading, not compiled, not run on hardware
//
#include <hip/hip_runtime.h>
#include <hip/hip_bf16.h>
#include <stdint.h>

typedef __attribute__((ext_vector_type(16))) _Float16 v16h;
typedef __attribute__((ext_vector_type(8)))  _Float16 v8h;
typedef __attribute__((ext_vector_type(8)))  float    v8f;
typedef __attribute__((ext_vector_type(4)))  int      v4i;

#define BN_EPS  1e-5f
#define KNN_K   20
#define NEG_INF (-3.4e38f)

#if __has_builtin(__builtin_amdgcn_global_load_async_to_lds_b128) && \
    __has_builtin(__builtin_amdgcn_s_wait_asynccnt)
#define HAVE_ASYNC_LDS 1
typedef __attribute__((address_space(1))) v4i* gptr_b128;
typedef __attribute__((address_space(3))) v4i* lptr_b128;
#else
#define HAVE_ASYNC_LDS 0
#endif

__device__ __forceinline__ float lrelu(float y) { return y >= 0.f ? y : 0.2f * y; }
__device__ __forceinline__ float bn_apply(float y, float g, float b, float m, float v) {
  return (y - m) * (g * __frsqrt_rn(v + BN_EPS)) + b;
}

// ------------------------------------------------------------------
// generic f32 -> f16 convert (for weights)
// ------------------------------------------------------------------
__global__ void f32_to_f16_kernel(const float* __restrict__ src,
                                  _Float16* __restrict__ dst, int n) {
  int i = blockIdx.x * 256 + threadIdx.x;
  if (i < n) dst[i] = (_Float16)src[i];
}

// ------------------------------------------------------------------
// KNN on raw 3D points (C=3): scalar, one thread per query point.
// x layout: (B, 3, N).  pd = -||xi - xj||^2, keep top-20 largest.
// ------------------------------------------------------------------
__global__ void knn3_kernel(const float* __restrict__ x, int* __restrict__ idx,
                            int B, int N) {
  int t = blockIdx.x * blockDim.x + threadIdx.x;
  if (t >= B * N) return;
  int b = t / N, i = t - b * N;
  const float* xb = x + (size_t)b * 3 * N;
  float xi = xb[i], yi = xb[N + i], zi = xb[2 * N + i];
  float best[KNN_K]; int bidx[KNN_K];
#pragma unroll
  for (int q = 0; q < KNN_K; ++q) { best[q] = NEG_INF; bidx[q] = 0; }
  for (int j = 0; j < N; ++j) {
    float dx = xb[j] - xi, dy = xb[N + j] - yi, dz = xb[2 * N + j] - zi;
    float pd = -(dx * dx + dy * dy + dz * dz);
    if (pd > best[KNN_K - 1]) {
      int p = KNN_K - 1;
      while (p > 0 && best[p - 1] < pd) {
        best[p] = best[p - 1]; bidx[p] = bidx[p - 1]; --p;
      }
      best[p] = pd; bidx[p] = j;
    }
  }
  for (int q = 0; q < KNN_K; ++q) idx[(size_t)t * KNN_K + q] = bidx[q];
}

// ------------------------------------------------------------------
// Stage 1: feat10 build + mask branch + conv1(6->64)+bn1+lrelu+max_k.
// One 64-thread block per (b, n).  Writes x1 into cat[..., 0:64].
// ------------------------------------------------------------------
__global__ __launch_bounds__(64)
void stage1_kernel(const float* __restrict__ x, const int* __restrict__ idx,
                   const float* __restrict__ W1,
                   const float* g1, const float* b1, const float* m1, const float* v1,
                   const float* __restrict__ Wm1,
                   const float* gm1, const float* bm1, const float* mm1, const float* vm1,
                   const float* __restrict__ Wm2,
                   const float* gm2, const float* bm2, const float* mm2, const float* vm2,
                   float* __restrict__ cat, float* __restrict__ maskB, int N) {
  int wg = blockIdx.x;
  int b = wg / N, n = wg - b * N;
  int tid = threadIdx.x;
  __shared__ float F[KNN_K][10];
  __shared__ float sM[KNN_K];
  const float* xb = x + (size_t)b * 3 * N;
  float cx = xb[n], cy = xb[N + n], cz = xb[2 * N + n];
  if (tid < KNN_K) {
    int j = idx[((size_t)b * N + n) * KNN_K + tid];
    float nx = xb[j], ny = xb[N + j], nz = xb[2 * N + j];
    float dx = nx - cx, dy = ny - cy, dz = nz - cz;
    float sq = dx * dx + dy * dy + dz * dz;
    float eu = (sq > 0.f) ? sqrtf(sq) : 0.f;
    float f[10] = {dx, dy, dz, cx, cy, cz, nx, ny, nz, eu};
#pragma unroll
    for (int c = 0; c < 10; ++c) F[tid][c] = f[c];
    // mask branch: 10->6 (bn+lrelu), 6->1 (bn+sigmoid)
    float m6[6];
#pragma unroll
    for (int u = 0; u < 6; ++u) {
      float acc = 0.f;
#pragma unroll
      for (int c = 0; c < 10; ++c) acc += Wm1[u * 10 + c] * f[c];
      m6[u] = lrelu(bn_apply(acc, gm1[u], bm1[u], mm1[u], vm1[u]));
    }
    float acc = 0.f;
#pragma unroll
    for (int u = 0; u < 6; ++u) acc += Wm2[u] * m6[u];
    float y = bn_apply(acc, gm2[0], bm2[0], mm2[0], vm2[0]);
    sM[tid] = 1.f / (1.f + __expf(-y));
  }
  __syncthreads();
  if (tid == 0) {
    float mx = sM[0];
    for (int k = 1; k < KNN_K; ++k) mx = fmaxf(mx, sM[k]);
    maskB[(size_t)b * N + n] = mx;
  }
  // conv1 on feat10[..., :6]
  int o = tid;  // 64 outputs
  float sc = g1[o] * __frsqrt_rn(v1[o] + BN_EPS);
  float me = m1[o], be = b1[o];
  float mx = NEG_INF;
  for (int k = 0; k < KNN_K; ++k) {
    float acc = 0.f;
#pragma unroll
    for (int c = 0; c < 6; ++c) acc += W1[o * 6 + c] * F[k][c];
    mx = fmaxf(mx, lrelu((acc - me) * sc + be));
  }
  cat[((size_t)b * N + n) * 512 + o] = mx;
}

// ------------------------------------------------------------------
// Convert a channel slice of cat (f32, stride 512) to compact f16
// feature buffer + per-point squared norms.  One 64-thread block/point.
// ------------------------------------------------------------------
__global__ __launch_bounds__(64)
void to_f16_norms_kernel(const float* __restrict__ cat, _Float16* __restrict__ featH,
                         float* __restrict__ norms, int srcOff, int C) {
  int p = blockIdx.x;  // b*N + n
  int tid = threadIdx.x;
  const float* src = cat + (size_t)p * 512 + srcOff;
  _Float16* dst = featH + (size_t)p * C;
  float s = 0.f;
  for (int c = tid; c < C; c += 64) {
    float v = src[c];
    dst[c] = (_Float16)v;
    s += v * v;
  }
  __shared__ float red[64];
  red[tid] = s;
  __syncthreads();
  for (int w = 32; w > 0; w >>= 1) {
    if (tid < w) red[tid] += red[tid + w];
    __syncthreads();
  }
  if (tid == 0) norms[p] = red[0];
}

// ------------------------------------------------------------------
// WMMA KNN: one wave per 16-row tile.  Gram tiles via v_wmma f16,
// pd = 2g - s_i - s_j, per-row top-20 maintained in LDS by lanes 0-15.
// ------------------------------------------------------------------
template <int C>
__global__ __launch_bounds__(32)
void knn_wmma_kernel(const _Float16* __restrict__ featH, const float* __restrict__ norms,
                     int* __restrict__ idx, int N) {
  constexpr int NCH = C / 32;
  int tilesPerB = N / 16;
  int wg = blockIdx.x;
  int b = wg / tilesPerB;
  int i0 = (wg - b * tilesPerB) * 16;
  int lane = threadIdx.x;
  int hi = lane >> 4, lr = lane & 15;

  __shared__ float sTile[16 * 16];
  __shared__ float sRowN[16];
  __shared__ float tv[16][KNN_K];
  __shared__ int   ti[16][KNN_K];
  if (lane < 16) {
    for (int q = 0; q < KNN_K; ++q) { tv[lane][q] = NEG_INF; ti[lane][q] = 0; }
    sRowN[lane] = norms[b * N + i0 + lane];
  }

  // preload A fragments for all channel chunks (rows i0..i0+15)
  const _Float16* fb = featH + (size_t)b * N * C;
  v16h a[NCH];
  {
    const _Float16* ap = fb + (size_t)(i0 + lr) * C;
    int kb = hi * 8;
#pragma unroll
    for (int ch = 0; ch < NCH; ++ch) {
      int c0 = ch * 32;
#pragma unroll
      for (int e = 0; e < 8; ++e) a[ch][e] = ap[c0 + kb + e];
#pragma unroll
      for (int e = 0; e < 8; ++e) a[ch][8 + e] = ap[c0 + 16 + kb + e];
    }
  }
  __syncthreads();

  for (int j0 = 0; j0 < N; j0 += 16) {
    v8f acc = {};
#pragma unroll
    for (int ch = 0; ch < NCH; ++ch) {
      v16h bf;
      const _Float16* bp = fb + (size_t)(j0 + lr) * C + ch * 32;
      __builtin_prefetch(bp + C * 16, 0, 1);
#pragma unroll
      for (int e = 0; e < 8; ++e) bf[e] = bp[hi * 8 + e];
#pragma unroll
      for (int e = 0; e < 8; ++e) bf[8 + e] = bp[16 + hi * 8 + e];
      acc = __builtin_amdgcn_wmma_f32_16x16x32_f16(false, a[ch], false, bf,
                                                   (short)0, acc, false, false);
    }
    float scol = norms[b * N + j0 + lr];  // this lane's column norm
#pragma unroll
    for (int r = 0; r < 8; ++r) {
      int row = r + 8 * hi;
      sTile[row * 16 + lr] = 2.0f * acc[r] - sRowN[row] - scol;
    }
    __syncthreads();
    if (lane < 16) {
      for (int c2 = 0; c2 < 16; ++c2) {
        float val = sTile[lane * 16 + c2];
        if (val > tv[lane][KNN_K - 1]) {
          int p = KNN_K - 1;
          while (p > 0 && tv[lane][p - 1] < val) {
            tv[lane][p] = tv[lane][p - 1]; ti[lane][p] = ti[lane][p - 1]; --p;
          }
          tv[lane][p] = val; ti[lane][p] = j0 + c2;
        }
      }
    }
    __syncthreads();
  }
  if (lane < 16) {
    size_t base = ((size_t)b * N + i0 + lane) * KNN_K;
    for (int q = 0; q < KNN_K; ++q) idx[base + q] = ti[lane][q];
  }
}

// ------------------------------------------------------------------
// WMMA EdgeConv: one 256-thread block per (b,n).  Builds the 32x2C
// f16 edge-feature tile in LDS (rows 20..31 duplicate k=0: max-safe),
// GEMMs against W (O x 2C, f16) with v_wmma, fused bn+lrelu epilogue,
// max over k, writes O channels into the cat slice.
// ------------------------------------------------------------------
template <int C, int O>
__global__ __launch_bounds__(256)
void edge_wmma_kernel(const _Float16* __restrict__ featH, const int* __restrict__ idx,
                      const _Float16* __restrict__ Wh,
                      const float* bg, const float* bb, const float* bm, const float* bv,
                      float* __restrict__ cat, int outOff, int N) {
  constexpr int C2 = 2 * C;
  constexpr int OT = O / 16;
  int wg = blockIdx.x;
  int b = wg / N, n = wg - b * N;
  int tid = threadIdx.x;
  int lane = tid & 31, wave = tid >> 5;
  int hi = lane >> 4, lr = lane & 15;

  __shared__ int sIdx[KNN_K];
  __shared__ _Float16 As[32 * C2];
  __shared__ float yS[KNN_K * O];

  if (tid < KNN_K) sIdx[tid] = idx[((size_t)b * N + n) * KNN_K + tid];
  __syncthreads();

  const _Float16* fb = featH + (size_t)b * N * C;
  for (int q = tid; q < 32 * C2; q += 256) {
    int m = q / C2, c = q - m * C2;
    int k = (m < KNN_K) ? m : 0;
    int j = sIdx[k];
    int cc = (c < C) ? c : (c - C);
    _Float16 ctr = fb[(size_t)n * C + cc];
    As[q] = (c < C) ? (_Float16)(fb[(size_t)j * C + cc] - ctr) : ctr;
  }
  __syncthreads();

  for (int t = wave; t < 2 * OT; t += 8) {
    int rt = t / OT;
    int ot = t - rt * OT;
    v8f acc = {};
#pragma unroll
    for (int ch = 0; ch < C2 / 32; ++ch) {
      v16h af, bf;
      const _Float16* ap = &As[(rt * 16 + lr) * C2 + ch * 32];
      int kb = hi * 8;
#pragma unroll
      for (int e = 0; e < 8; ++e) af[e] = ap[kb + e];
#pragma unroll
      for (int e = 0; e < 8; ++e) af[8 + e] = ap[16 + kb + e];
      const _Float16* bp = Wh + (size_t)(ot * 16 + lr) * C2 + ch * 32;
      __builtin_prefetch(bp + 32, 0, 1);
#pragma unroll
      for (int e = 0; e < 8; ++e) bf[e] = bp[hi * 8 + e];
#pragma unroll
      for (int e = 0; e < 8; ++e) bf[8 + e] = bp[16 + hi * 8 + e];
      acc = __builtin_amdgcn_wmma_f32_16x16x32_f16(false, af, false, bf,
                                                   (short)0, acc, false, false);
    }
    int o = ot * 16 + lr;
    float sc = bg[o] * __frsqrt_rn(bv[o] + BN_EPS);
    float me = bm[o], be = bb[o];
#pragma unroll
    for (int r = 0; r < 8; ++r) {
      int m = rt * 16 + r + 8 * hi;
      if (m < KNN_K) yS[m * O + o] = lrelu((acc[r] - me) * sc + be);
    }
  }
  __syncthreads();
  for (int o = tid; o < O; o += 256) {
    float mx = yS[o];
    for (int m = 1; m < KNN_K; ++m) mx = fmaxf(mx, yS[m * O + o]);
    cat[((size_t)b * N + n) * 512 + outOff + o] = mx;
  }
}

// ------------------------------------------------------------------
// Global pooling buffers init
// ------------------------------------------------------------------
__global__ void init_pool_kernel(unsigned* gmax, float* gsum, int n) {
  int i = blockIdx.x * 256 + threadIdx.x;
  if (i < n) { gmax[i] = 0u; gsum[i] = 0.f; }
}

// ------------------------------------------------------------------
// W5 (512->1024) WMMA GEMM with fused bn5+lrelu+mask+relu epilogue
// and atomic max/sum pooling over N.  One block per 16 points.
// A tile staged to LDS with GLOBAL_LOAD_ASYNC_TO_LDS_B128 when the
// async builtins are available (CDNA5 async path, ASYNCcnt).
// ------------------------------------------------------------------
__global__ __launch_bounds__(256)
void w5_pool_kernel(const _Float16* __restrict__ catH, const _Float16* __restrict__ W5h,
                    const float* bg, const float* bb, const float* bm, const float* bv,
                    const float* __restrict__ maskB,
                    unsigned* __restrict__ gmax, float* __restrict__ gsum, int N) {
  int p0 = blockIdx.x * 16;  // global row (b*N + n)
  int b = p0 / N;
  int tid = threadIdx.x;
  int lane = tid & 31, wave = tid >> 5;
  int hi = lane >> 4, lr = lane & 15;

  __shared__ _Float16 As[16 * 512];
  __shared__ float sMask[16];

  const _Float16* gsrc = catH + (size_t)p0 * 512;
#if HAVE_ASYNC_LDS
  {
    // 16*512 halfs = 16 KB; 256 threads x 16 B = 4 KB per pass, 4 passes.
#pragma unroll
    for (int q8 = 0; q8 < 16 * 512; q8 += 256 * 8) {
      int e = q8 + tid * 8;
      __builtin_amdgcn_global_load_async_to_lds_b128(
          (gptr_b128)(gsrc + e), (lptr_b128)(&As[e]), 0, 0);
    }
    __builtin_amdgcn_s_wait_asynccnt(0);
  }
#else
  for (int q8 = tid * 8; q8 < 16 * 512; q8 += 256 * 8) {
    *(v8h*)(&As[q8]) = *(const v8h*)(gsrc + q8);
  }
#endif
  if (tid < 16) sMask[tid] = maskB[p0 + tid];
  __syncthreads();

  for (int ot = wave; ot < 64; ot += 8) {
    v8f acc = {};
#pragma unroll 4
    for (int ch = 0; ch < 16; ++ch) {
      v16h af, bf;
      const _Float16* ap = &As[lr * 512 + ch * 32];
      int kb = hi * 8;
#pragma unroll
      for (int e = 0; e < 8; ++e) af[e] = ap[kb + e];
#pragma unroll
      for (int e = 0; e < 8; ++e) af[8 + e] = ap[16 + kb + e];
      const _Float16* bp = W5h + (size_t)(ot * 16 + lr) * 512 + ch * 32;
      __builtin_prefetch(bp + 32, 0, 1);
#pragma unroll
      for (int e = 0; e < 8; ++e) bf[e] = bp[hi * 8 + e];
#pragma unroll
      for (int e = 0; e < 8; ++e) bf[8 + e] = bp[16 + hi * 8 + e];
      acc = __builtin_amdgcn_wmma_f32_16x16x32_f16(false, af, false, bf,
                                                   (short)0, acc, false, false);
    }
    int o = ot * 16 + lr;
    float sc = bg[o] * __frsqrt_rn(bv[o] + BN_EPS);
    float me = bm[o], be = bb[o];
    float lmax = 0.f, lsum = 0.f;
#pragma unroll
    for (int r = 0; r < 8; ++r) {
      int m = r + 8 * hi;
      float y = lrelu((acc[r] - me) * sc + be);
      y = fmaxf(y * sMask[m], 0.f);   // mask then relu (post-relu >= 0)
      lmax = fmaxf(lmax, y);
      lsum += y;
    }
    atomicMax(&gmax[b * 1024 + o], __float_as_uint(lmax));
    atomicAdd(&gsum[b * 1024 + o], lsum);
  }
}

__global__ void finalize_pool_kernel(const unsigned* __restrict__ gmax,
                                     const float* __restrict__ gsum,
                                     float* __restrict__ g, int N, int n) {
  int i = blockIdx.x * 256 + threadIdx.x;
  if (i < n) {
    int b = i / 1024, o = i - b * 1024;
    g[b * 2048 + o] = __uint_as_float(gmax[i]);
    g[b * 2048 + 1024 + o] = gsum[i] / (float)N;
  }
}

// ------------------------------------------------------------------
// Head MLP (B=8): 2048->512 (bn6,lrelu) -> 256 (+b2,bn7,lrelu) -> 40 (+b3)
// One 512-thread block per batch element; g staged in LDS.
// ------------------------------------------------------------------
__global__ __launch_bounds__(512)
void head_kernel(const float* __restrict__ g,
                 const float* __restrict__ L1,
                 const float* g6, const float* b6, const float* m6, const float* v6,
                 const float* __restrict__ L2, const float* __restrict__ bias2,
                 const float* g7, const float* b7, const float* m7, const float* v7,
                 const float* __restrict__ L3, const float* __restrict__ bias3,
                 float* __restrict__ out) {
  int b = blockIdx.x;
  int tid = threadIdx.x;
  __shared__ float sg[2048];
  __shared__ float h1[512];
  __shared__ float h2[256];
  const float* gv = g + (size_t)b * 2048;
#pragma unroll
  for (int c = tid; c < 2048; c += 512) sg[c] = gv[c];
  __syncthreads();
  {
    float acc = 0.f;
    const float* w = L1 + (size_t)tid * 2048;
    for (int c = 0; c < 2048; ++c) acc += w[c] * sg[c];
    h1[tid] = lrelu(bn_apply(acc, g6[tid], b6[tid], m6[tid], v6[tid]));
  }
  __syncthreads();
  if (tid < 256) {
    float acc = bias2[tid];
    const float* w = L2 + (size_t)tid * 512;
    for (int c = 0; c < 512; ++c) acc += w[c] * h1[c];
    h2[tid] = lrelu(bn_apply(acc, g7[tid], b7[tid], m7[tid], v7[tid]));
  }
  __syncthreads();
  if (tid < 40) {
    float acc = bias3[tid];
    const float* w = L3 + (size_t)tid * 256;
    for (int c = 0; c < 256; ++c) acc += w[c] * h2[c];
    out[b * 40 + tid] = acc;
  }
}

// ------------------------------------------------------------------
// launch
// ------------------------------------------------------------------
extern "C" void kernel_launch(void* const* d_in, const int* in_sizes, int n_in,
                              void* d_out, int out_size, void* d_ws, size_t ws_size,
                              hipStream_t stream) {
  (void)n_in; (void)out_size; (void)ws_size;
  const int B = 8;
  const int N = in_sizes[0] / (3 * B);  // 2048

  const float* x   = (const float*)d_in[0];
  const float* W1  = (const float*)d_in[1];
  const float* bn1g = (const float*)d_in[2],  *bn1b = (const float*)d_in[3],
             * bn1m = (const float*)d_in[4],  *bn1v = (const float*)d_in[5];
  const float* W2  = (const float*)d_in[6];
  const float* bn2g = (const float*)d_in[7],  *bn2b = (const float*)d_in[8],
             * bn2m = (const float*)d_in[9],  *bn2v = (const float*)d_in[10];
  const float* W3  = (const float*)d_in[11];
  const float* bn3g = (const float*)d_in[12], *bn3b = (const float*)d_in[13],
             * bn3m = (const float*)d_in[14], *bn3v = (const float*)d_in[15];
  const float* W4  = (const float*)d_in[16];
  const float* bn4g = (const float*)d_in[17], *bn4b = (const float*)d_in[18],
             * bn4m = (const float*)d_in[19], *bn4v = (const float*)d_in[20];
  const float* W5  = (const float*)d_in[21];
  const float* bn5g = (const float*)d_in[22], *bn5b = (const float*)d_in[23],
             * bn5m = (const float*)d_in[24], *bn5v = (const float*)d_in[25];
  const float* Wm1 = (const float*)d_in[26];
  const float* bnm1g = (const float*)d_in[27], *bnm1b = (const float*)d_in[28],
             * bnm1m = (const float*)d_in[29], *bnm1v = (const float*)d_in[30];
  const float* Wm2 = (const float*)d_in[31];
  const float* bnm2g = (const float*)d_in[32], *bnm2b = (const float*)d_in[33],
             * bnm2m = (const float*)d_in[34], *bnm2v = (const float*)d_in[35];
  const float* L1  = (const float*)d_in[36];
  const float* bn6g = (const float*)d_in[37], *bn6b = (const float*)d_in[38],
             * bn6m = (const float*)d_in[39], *bn6v = (const float*)d_in[40];
  const float* L2  = (const float*)d_in[41];
  const float* b2  = (const float*)d_in[42];
  const float* bn7g = (const float*)d_in[43], *bn7b = (const float*)d_in[44],
             * bn7m = (const float*)d_in[45], *bn7v = (const float*)d_in[46];
  const float* L3  = (const float*)d_in[47];
  const float* b3  = (const float*)d_in[48];

  // ---- workspace bump allocation ----
  uintptr_t base = (uintptr_t)d_ws;
  size_t off = 0;
  auto take = [&](size_t bytes) -> uintptr_t {
    uintptr_t p = base + off;
    off += (bytes + 255) & ~(size_t)255;
    return p;
  };
  int*      idxB  = (int*)      take((size_t)B * N * KNN_K * 4);
  float*    cat   = (float*)    take((size_t)B * N * 512 * 4);
  _Float16* featH = (_Float16*) take((size_t)B * N * 512 * 2);
  float*    norms = (float*)    take((size_t)B * N * 4);
  float*    maskB = (float*)    take((size_t)B * N * 4);
  _Float16* W2h   = (_Float16*) take((size_t)64 * 128 * 2);
  _Float16* W3h   = (_Float16*) take((size_t)128 * 128 * 2);
  _Float16* W4h   = (_Float16*) take((size_t)256 * 256 * 2);
  _Float16* W5h   = (_Float16*) take((size_t)1024 * 512 * 2);
  unsigned* gmax  = (unsigned*) take((size_t)B * 1024 * 4);
  float*    gsum  = (float*)    take((size_t)B * 1024 * 4);
  float*    gbuf  = (float*)    take((size_t)B * 2048 * 4);

  // ---- weight conversion ----
  f32_to_f16_kernel<<<(64 * 128 + 255) / 256, 256, 0, stream>>>(W2, W2h, 64 * 128);
  f32_to_f16_kernel<<<(128 * 128 + 255) / 256, 256, 0, stream>>>(W3, W3h, 128 * 128);
  f32_to_f16_kernel<<<(256 * 256 + 255) / 256, 256, 0, stream>>>(W4, W4h, 256 * 256);
  f32_to_f16_kernel<<<(1024 * 512 + 255) / 256, 256, 0, stream>>>(W5, W5h, 1024 * 512);

  // ---- stage 1 ----
  knn3_kernel<<<(B * N + 255) / 256, 256, 0, stream>>>(x, idxB, B, N);
  stage1_kernel<<<B * N, 64, 0, stream>>>(x, idxB,
      W1, bn1g, bn1b, bn1m, bn1v,
      Wm1, bnm1g, bnm1b, bnm1m, bnm1v,
      Wm2, bnm2g, bnm2b, bnm2m, bnm2v,
      cat, maskB, N);

  // ---- stage 2 (x1: C=64 -> O=64, cat[64:128)) ----
  to_f16_norms_kernel<<<B * N, 64, 0, stream>>>(cat, featH, norms, 0, 64);
  knn_wmma_kernel<64><<<B * N / 16, 32, 0, stream>>>(featH, norms, idxB, N);
  edge_wmma_kernel<64, 64><<<B * N, 256, 0, stream>>>(featH, idxB, W2h,
      bn2g, bn2b, bn2m, bn2v, cat, 64, N);

  // ---- stage 3 (x2: C=64 -> O=128, cat[128:256)) ----
  to_f16_norms_kernel<<<B * N, 64, 0, stream>>>(cat, featH, norms, 64, 64);
  knn_wmma_kernel<64><<<B * N / 16, 32, 0, stream>>>(featH, norms, idxB, N);
  edge_wmma_kernel<64, 128><<<B * N, 256, 0, stream>>>(featH, idxB, W3h,
      bn3g, bn3b, bn3m, bn3v, cat, 128, N);

  // ---- stage 4 (x3: C=128 -> O=256, cat[256:512)) ----
  to_f16_norms_kernel<<<B * N, 64, 0, stream>>>(cat, featH, norms, 128, 128);
  knn_wmma_kernel<128><<<B * N / 16, 32, 0, stream>>>(featH, norms, idxB, N);
  edge_wmma_kernel<128, 256><<<B * N, 256, 0, stream>>>(featH, idxB, W4h,
      bn4g, bn4b, bn4m, bn4v, cat, 256, N);

  // ---- W5 + mask + pooling ----
  to_f16_norms_kernel<<<B * N, 64, 0, stream>>>(cat, featH, norms, 0, 512);
  init_pool_kernel<<<(B * 1024 + 255) / 256, 256, 0, stream>>>(gmax, gsum, B * 1024);
  w5_pool_kernel<<<B * N / 16, 256, 0, stream>>>(featH, W5h,
      bn5g, bn5b, bn5m, bn5v, maskB, gmax, gsum, N);
  finalize_pool_kernel<<<(B * 1024 + 255) / 256, 256, 0, stream>>>(gmax, gsum, gbuf, N, B * 1024);

  // ---- head ----
  head_kernel<<<B, 512, 0, stream>>>(gbuf,
      L1, bn6g, bn6b, bn6m, bn6v,
      L2, b2, bn7g, bn7b, bn7m, bn7v,
      L3, b3, (float*)d_out);
}
